// Lfm2MoeShortConvDecoderLayer_2113123909698
// MI455X (gfx1250) — compile-verified, matrix operand order checked by reading, not run on hardware
//
#include <hip/hip_runtime.h>
#include <hip/hip_bf16.h>

// ---------------- problem constants ----------------
#define T_TOK 2048
#define DHID  1024
#define DH3   3072
#define FFH   512
#define NEXP  16
#define TOPK  4
#define LDSH  (DHID + 8)   // padded LDS row stride (halves) for h tile
#define LDSA  (FFH + 8)    // padded LDS row stride (halves) for act tile

typedef __attribute__((ext_vector_type(16))) __bf16 v16bf;
typedef __attribute__((ext_vector_type(8)))  float  v8f;

// Load the 16 bf16 elements a lane needs for a WMMA A/B operand:
// two contiguous 16-byte chunks (K = k+kb..k+kb+7 and K = k+kb+16..k+kb+23).
__device__ __forceinline__ v16bf ld2u(const void* p0, const void* p1) {
  union { v16bf v; uint4 u[2]; } t;
  t.u[0] = *reinterpret_cast<const uint4*>(p0);
  t.u[1] = *reinterpret_cast<const uint4*>(p1);
  return t.v;
}

__device__ __forceinline__ v8f wmma_bf16(v16bf a, v16bf b, v8f c) {
  return __builtin_amdgcn_wmma_f32_16x16x32_bf16(false, a, false, b, (short)0, c,
                                                 false, false);
}

__device__ __forceinline__ unsigned short f2bf_bits(float f) {
  __hip_bfloat16 b = __float2bfloat16(f);
  return *reinterpret_cast<unsigned short*>(&b);
}

// ---------------- fp32 -> bf16 bulk convert (4 elems/thread) ----------------
__global__ __launch_bounds__(256) void cvt4_kernel(const float4* __restrict__ src,
                                                   uint2* __restrict__ dst, int n4) {
  int i = blockIdx.x * 256 + threadIdx.x;
  if (i >= n4) return;
  float4 f = src[i];
  unsigned int a = (unsigned int)f2bf_bits(f.x) | ((unsigned int)f2bf_bits(f.y) << 16);
  unsigned int b = (unsigned int)f2bf_bits(f.z) | ((unsigned int)f2bf_bits(f.w) << 16);
  dst[i] = make_uint2(a, b);
}

// ---------------- fused add + RMSNorm (one block per token) ----------------
__global__ __launch_bounds__(256) void norm_kernel(const float* __restrict__ x,
                                                   const float* __restrict__ res_in,
                                                   const float* __restrict__ w,
                                                   __hip_bfloat16* __restrict__ h_out,
                                                   float* __restrict__ res_out) {
  const int t = blockIdx.x;
  const float* xr = x + (size_t)t * DHID;
  const float* rr = res_in + (size_t)t * DHID;
  float v[4];
  float ss = 0.f;
#pragma unroll
  for (int i = 0; i < 4; ++i) {
    int d = threadIdx.x + i * 256;
    float a = xr[d] + rr[d];
    v[i] = a;
    ss += a * a;
  }
  __shared__ float red[8];
  __shared__ float srs;
#pragma unroll
  for (int o = 16; o > 0; o >>= 1) ss += __shfl_down(ss, o, 32);
  if ((threadIdx.x & 31) == 0) red[threadIdx.x >> 5] = ss;
  __syncthreads();
  if (threadIdx.x == 0) {
    float s = 0.f;
    for (int i = 0; i < 8; ++i) s += red[i];
    srs = rsqrtf(s / (float)DHID + 1e-5f);
  }
  __syncthreads();
  float rs = srs;
#pragma unroll
  for (int i = 0; i < 4; ++i) {
    int d = threadIdx.x + i * 256;
    res_out[(size_t)t * DHID + d] = v[i];
    h_out[(size_t)t * DHID + d] = __float2bfloat16(v[i] * rs * w[d]);
  }
}

// ---------------- generic bf16 WMMA GEMM: C[M,N] = A[M,K] * W[N,K]^T ----------
// One wave per 32x64 output block (2 M-tiles x 4 N-tiles, K-step 32).
__global__ __launch_bounds__(256) void gemm_bf16_kernel(
    const __hip_bfloat16* __restrict__ A, int lda,
    const __hip_bfloat16* __restrict__ W, int ldw,
    float* __restrict__ C, int ldc, int M, int N, int K) {
  const int wave = blockIdx.x * 8 + (threadIdx.x >> 5);
  const int tn = N >> 6, tm = M >> 5;
  if (wave >= tm * tn) return;
  const int wm = wave / tn, wn = wave % tn;
  const int lane = threadIdx.x & 31;
  const int r = lane & 15, hi = lane >> 4;
  const int kb = hi << 3;
  const size_t m0 = (size_t)wm << 5, n0 = (size_t)wn << 6;

  const __hip_bfloat16* a0 = A + (m0 + r) * (size_t)lda;
  const __hip_bfloat16* a1 = A + (m0 + 16 + r) * (size_t)lda;
  const __hip_bfloat16* b0 = W + (n0 + r) * (size_t)ldw;
  const __hip_bfloat16* b1 = W + (n0 + 16 + r) * (size_t)ldw;
  const __hip_bfloat16* b2 = W + (n0 + 32 + r) * (size_t)ldw;
  const __hip_bfloat16* b3 = W + (n0 + 48 + r) * (size_t)ldw;

  v8f acc[2][4] = {};
  for (int k = 0; k < K; k += 32) {
    const int o0 = k + kb, o1 = o0 + 16;
    v16bf A0 = ld2u(a0 + o0, a0 + o1);
    v16bf A1 = ld2u(a1 + o0, a1 + o1);
    v16bf B0 = ld2u(b0 + o0, b0 + o1);
    v16bf B1 = ld2u(b1 + o0, b1 + o1);
    v16bf B2 = ld2u(b2 + o0, b2 + o1);
    v16bf B3 = ld2u(b3 + o0, b3 + o1);
    acc[0][0] = wmma_bf16(A0, B0, acc[0][0]);
    acc[0][1] = wmma_bf16(A0, B1, acc[0][1]);
    acc[0][2] = wmma_bf16(A0, B2, acc[0][2]);
    acc[0][3] = wmma_bf16(A0, B3, acc[0][3]);
    acc[1][0] = wmma_bf16(A1, B0, acc[1][0]);
    acc[1][1] = wmma_bf16(A1, B1, acc[1][1]);
    acc[1][2] = wmma_bf16(A1, B2, acc[1][2]);
    acc[1][3] = wmma_bf16(A1, B3, acc[1][3]);
  }
  const int n = lane & 15;
#pragma unroll
  for (int i = 0; i < 2; ++i)
#pragma unroll
    for (int j = 0; j < 4; ++j)
#pragma unroll
      for (int v = 0; v < 8; ++v)
        C[(m0 + i * 16 + v + hi * 8) * (size_t)ldc + n0 + j * 16 + n] = acc[i][j][v];
}

// ---------------- Bx = B*x, causal depthwise conv(L=3), gate by C -> bf16 ----
__global__ __launch_bounds__(256) void conv_gate_kernel(const float* __restrict__ bcx,
                                                        const float* __restrict__ cw,
                                                        __hip_bfloat16* __restrict__ gated) {
  const int idx = blockIdx.x * 256 + threadIdx.x;  // exactly T*D threads
  const int t = idx >> 10, d = idx & 1023;
  const float* row = bcx + (size_t)t * DH3;
  const float w0 = cw[d * 3 + 0], w1 = cw[d * 3 + 1], w2 = cw[d * 3 + 2];
  float s = w2 * (row[d] * row[2048 + d]);                     // Bx[t]
  if (t >= 1) s += w1 * (row[d - DH3] * row[2048 + d - DH3]);  // Bx[t-1]
  if (t >= 2) s += w0 * (row[d - 2 * DH3] * row[2048 + d - 2 * DH3]);
  gated[idx] = __float2bfloat16(row[1024 + d] * s);
}

// ---------------- routing: sigmoid + biased top-4, scatter to expert lists ---
__global__ __launch_bounds__(256) void route_kernel(const __hip_bfloat16* __restrict__ h,
                                                    const float* __restrict__ gate_w,
                                                    const float* __restrict__ gate_bias,
                                                    int* __restrict__ counts,
                                                    int* __restrict__ tok_ids,
                                                    float* __restrict__ tok_w) {
  const int wave = threadIdx.x >> 5, lane = threadIdx.x & 31;
  const int t = blockIdx.x * 8 + wave;
  const int e = lane & 15, half = lane >> 4;
  const __hip_bfloat16* hr = h + (size_t)t * DHID + half * 512;
  const float* gr = gate_w + (size_t)e * DHID + half * 512;
  float acc = 0.f;
  for (int d = 0; d < 512; ++d) acc += __bfloat162float(hr[d]) * gr[d];
  acc += __shfl_xor(acc, 16, 32);  // full dot in every lane
  const float score = 1.f / (1.f + __expf(-acc));
  const float choice = score + gate_bias[e];

  float bv[4] = {-1e30f, -1e30f, -1e30f, -1e30f};
  float bs[4] = {0.f, 0.f, 0.f, 0.f};
  int bi[4] = {0, 0, 0, 0};
  for (int q = 0; q < NEXP; ++q) {
    float cq = __shfl(choice, q, 32);
    float sq = __shfl(score, q, 32);
    for (int k = 0; k < TOPK; ++k) {
      if (cq > bv[k]) {
        for (int j = TOPK - 1; j > k; --j) { bv[j] = bv[j-1]; bs[j] = bs[j-1]; bi[j] = bi[j-1]; }
        bv[k] = cq; bs[k] = sq; bi[k] = q;
        break;
      }
    }
  }
  const float inv = 1.f / (bs[0] + bs[1] + bs[2] + bs[3] + 1e-20f);
  if (lane == 0) {
    for (int k = 0; k < TOPK; ++k) {
      int ee = bi[k];
      int slot = atomicAdd(counts + ee, 1);
      tok_ids[ee * T_TOK + slot] = t;
      tok_w[ee * T_TOK + slot] = bs[k] * inv;  // * SCALE (=1.0)
    }
  }
}

// ---------------- grouped gathered MoE tile: one block per (expert, 16 toks) -
__global__ __launch_bounds__(256) void moe_kernel(const __hip_bfloat16* __restrict__ h,
                                                  const __hip_bfloat16* __restrict__ w1,
                                                  const __hip_bfloat16* __restrict__ w2,
                                                  const int* __restrict__ counts,
                                                  const int* __restrict__ tok_ids,
                                                  const float* __restrict__ tok_w,
                                                  float* __restrict__ out) {
  const int e = blockIdx.x >> 7;       // 128 tiles per expert
  const int tile = blockIdx.x & 127;
  const int cnt = counts[e];
  const int m0 = tile << 4;
  if (m0 >= cnt) return;

  __shared__ unsigned short sh_h[16 * LDSH];   // gathered h rows (bf16)
  __shared__ unsigned short sh_a[16 * LDSA];   // silu(g)*u activations (bf16)
  __shared__ int   sh_tok[16];
  __shared__ float sh_w[16];

  const int tid = threadIdx.x;
  if (tid < 16) {
    int row = m0 + tid;
    bool valid = row < cnt;
    sh_tok[tid] = valid ? tok_ids[e * T_TOK + row] : 0;
    sh_w[tid]   = valid ? tok_w[e * T_TOK + row] : 0.f;
  }
  __syncthreads();

  // gather 16 token rows (16 x 1024 bf16 = 32KB) into padded LDS, 16B granules.
  // On gfx1250 use the async VMEM->LDS path (no VGPR round-trip, ASYNCcnt).
#if defined(__gfx1250__)
  {
    const unsigned lds_base = (unsigned)(unsigned long long)(&sh_h[0]);
    for (int i = tid; i < 16 * 128; i += 256) {
      int row = i >> 7, c = i & 127;
      unsigned long long g = (unsigned long long)(h + (size_t)sh_tok[row] * DHID) +
                             (unsigned long long)c * 16u;
      unsigned l = lds_base + (unsigned)(row * (LDSH * 2) + c * 16);
      asm volatile("global_load_async_to_lds_b128 %0, %1, off"
                   :: "v"(l), "v"(g) : "memory");
    }
    asm volatile("s_wait_asynccnt 0x0" ::: "memory");
  }
#else
  for (int i = tid; i < 16 * 128; i += 256) {
    int row = i >> 7, c = i & 127;
    const uint4* src = reinterpret_cast<const uint4*>(h + (size_t)sh_tok[row] * DHID) + c;
    reinterpret_cast<uint4*>(sh_h)[row * (LDSH / 8) + c] = *src;
  }
#endif
  __syncthreads();

  const int wave = tid >> 5, lane = tid & 31;
  const int r = lane & 15, hi = lane >> 4;
  const int kb = hi << 3, n = lane & 15;
  const unsigned short* ah = sh_h + r * LDSH;

  // Phase A: gu = h_tile * w1[e]^T ; act = silu(g) * u  (32 f-tiles / 8 waves)
  const __hip_bfloat16* w1e = w1 + (size_t)e * (2 * FFH) * DHID;
  for (int ft = wave; ft < 32; ft += 8) {
    const __hip_bfloat16* bg = w1e + (size_t)(ft * 16 + r) * DHID;
    const __hip_bfloat16* bu = w1e + (size_t)(FFH + ft * 16 + r) * DHID;
    v8f ag = {}, au = {};
    for (int k = 0; k < DHID; k += 32) {
      const int o0 = k + kb, o1 = o0 + 16;
      v16bf Aa = ld2u(ah + o0, ah + o1);
      v16bf Bg = ld2u(bg + o0, bg + o1);
      v16bf Bu = ld2u(bu + o0, bu + o1);
      ag = wmma_bf16(Aa, Bg, ag);
      au = wmma_bf16(Aa, Bu, au);
    }
#pragma unroll
    for (int v = 0; v < 8; ++v) {
      float g = ag[v];
      float a = (g / (1.f + __expf(-g))) * au[v];
      sh_a[(v + hi * 8) * LDSA + ft * 16 + n] = f2bf_bits(a);
    }
  }
  __syncthreads();

  // Phase B: ye = act * w2[e]^T, weighted atomic accumulate (64 d-tiles / 8 waves)
  const __hip_bfloat16* w2e = w2 + (size_t)e * DHID * FFH;
  const unsigned short* aa = sh_a + r * LDSA;
  for (int dt = wave; dt < 64; dt += 8) {
    const __hip_bfloat16* bw = w2e + (size_t)(dt * 16 + r) * FFH;
    v8f acc = {};
    for (int k = 0; k < FFH; k += 32) {
      const int o0 = k + kb, o1 = o0 + 16;
      v16bf Aa = ld2u(aa + o0, aa + o1);
      v16bf Bw = ld2u(bw + o0, bw + o1);
      acc = wmma_bf16(Aa, Bw, acc);
    }
#pragma unroll
    for (int v = 0; v < 8; ++v) {
      int m = v + hi * 8;
      atomicAdd(out + (size_t)sh_tok[m] * DHID + dt * 16 + n, acc[v] * sh_w[m]);
    }
  }
}

// ---------------- host orchestration ----------------
static inline char* carve(char*& p, size_t bytes) {
  char* q = p;
  p += (bytes + 255) & ~(size_t)255;
  return q;
}

extern "C" void kernel_launch(void* const* d_in, const int* in_sizes, int n_in,
                              void* d_out, int out_size, void* d_ws, size_t ws_size,
                              hipStream_t stream) {
  const float* hidden    = (const float*)d_in[0];
  const float* residual  = (const float*)d_in[1];
  const float* op_norm_w = (const float*)d_in[2];
  const float* ffn_norm_w= (const float*)d_in[3];
  const float* in_proj_w = (const float*)d_in[4];   // [3072,1024]
  const float* conv_w    = (const float*)d_in[5];   // [1024,3]
  const float* out_proj_w= (const float*)d_in[6];   // [1024,1024]
  const float* gate_w    = (const float*)d_in[7];   // [16,1024]
  const float* gate_bias = (const float*)d_in[8];   // [16]
  const float* w1        = (const float*)d_in[9];   // [16,1024,1024]
  const float* w2        = (const float*)d_in[10];  // [16,1024,512]
  float* out = (float*)d_out;                       // [0,T*D)=out, [T*D,2T*D)=residual

  char* p = (char*)d_ws;
  __hip_bfloat16* h1_bf  = (__hip_bfloat16*)carve(p, (size_t)T_TOK * DHID * 2);
  float*          res1   = (float*)         carve(p, (size_t)T_TOK * DHID * 4);
  float*          bcx    = (float*)         carve(p, (size_t)T_TOK * DH3 * 4);
  __hip_bfloat16* gated  = (__hip_bfloat16*)carve(p, (size_t)T_TOK * DHID * 2);
  float*          y      = (float*)         carve(p, (size_t)T_TOK * DHID * 4);
  __hip_bfloat16* h2_bf  = (__hip_bfloat16*)carve(p, (size_t)T_TOK * DHID * 2);
  __hip_bfloat16* ipw_bf = (__hip_bfloat16*)carve(p, (size_t)DH3 * DHID * 2);
  __hip_bfloat16* opw_bf = (__hip_bfloat16*)carve(p, (size_t)DHID * DHID * 2);
  __hip_bfloat16* w1_bf  = (__hip_bfloat16*)carve(p, (size_t)NEXP * 2 * FFH * DHID * 2);
  __hip_bfloat16* w2_bf  = (__hip_bfloat16*)carve(p, (size_t)NEXP * DHID * FFH * 2);
  int*            counts = (int*)           carve(p, (size_t)NEXP * 4);
  int*            tok_ids= (int*)           carve(p, (size_t)NEXP * T_TOK * 4);
  float*          tok_w  = (float*)         carve(p, (size_t)NEXP * T_TOK * 4);

  // zero MoE output accumulator + expert counters (graph-capture safe)
  hipMemsetAsync(out, 0, (size_t)T_TOK * DHID * 4, stream);
  hipMemsetAsync(counts, 0, NEXP * 4, stream);

  // weight conversions fp32 -> bf16
  {
    int n4;
    n4 = DH3 * DHID / 4;
    cvt4_kernel<<<(n4 + 255) / 256, 256, 0, stream>>>((const float4*)in_proj_w, (uint2*)ipw_bf, n4);
    n4 = DHID * DHID / 4;
    cvt4_kernel<<<(n4 + 255) / 256, 256, 0, stream>>>((const float4*)out_proj_w, (uint2*)opw_bf, n4);
    n4 = NEXP * 2 * FFH * DHID / 4;
    cvt4_kernel<<<(n4 + 255) / 256, 256, 0, stream>>>((const float4*)w1, (uint2*)w1_bf, n4);
    n4 = NEXP * DHID * FFH / 4;
    cvt4_kernel<<<(n4 + 255) / 256, 256, 0, stream>>>((const float4*)w2, (uint2*)w2_bf, n4);
  }

  // h1, res1 = fused_add_rmsnorm(hidden, op_norm_w, residual)
  norm_kernel<<<T_TOK, 256, 0, stream>>>(hidden, residual, op_norm_w, h1_bf, res1);

  // bcx = h1 @ in_proj_w^T   [2048 x 3072]
  {
    int waves = (T_TOK / 32) * (DH3 / 64);
    gemm_bf16_kernel<<<waves / 8, 256, 0, stream>>>(h1_bf, DHID, ipw_bf, DHID,
                                                    bcx, DH3, T_TOK, DH3, DHID);
  }

  // gated = C * causal_conv(B*x)  -> bf16
  conv_gate_kernel<<<T_TOK * DHID / 256, 256, 0, stream>>>(bcx, conv_w, gated);

  // y = gated @ out_proj_w^T   [2048 x 1024]
  {
    int waves = (T_TOK / 32) * (DHID / 64);
    gemm_bf16_kernel<<<waves / 8, 256, 0, stream>>>(gated, DHID, opw_bf, DHID,
                                                    y, DHID, T_TOK, DHID, DHID);
  }

  // h2, residual_out = fused_add_rmsnorm(y, ffn_norm_w, res1); residual -> d_out 2nd half
  norm_kernel<<<T_TOK, 256, 0, stream>>>(y, res1, ffn_norm_w, h2_bf,
                                         out + (size_t)T_TOK * DHID);

  // routing: top-4 per token into per-expert token lists
  route_kernel<<<T_TOK / 8, 256, 0, stream>>>(h2_bf, gate_w, gate_bias,
                                              counts, tok_ids, tok_w);

  // sparse MoE grouped GEMM with weighted atomic combine into out
  moe_kernel<<<NEXP * (T_TOK / 16), 256, 0, stream>>>(h2_bf, w1_bf, w2_bf,
                                                      counts, tok_ids, tok_w, out);

  (void)in_sizes; (void)n_in; (void)out_size; (void)ws_size;
}
